// BahdanauAttention_326417515272
// MI455X (gfx1250) — matrix operand order repaired and checked by reading, not compiled, not run on hardware
//
#include <hip/hip_runtime.h>
#include <hip/hip_bf16.h>

// Problem constants (match reference)
#define BB      32
#define S_LEN   2048
#define DIM     1024          // IN_DIM == OUT_DIM == 1024
#define APAD    1032          // padded LDS row stride (bf16 elems), 16B-aligned rows

typedef __attribute__((ext_vector_type(16))) __bf16 v16bf;
typedef __attribute__((ext_vector_type(8)))  float  v8f;

union FragB16 {
    v16bf h;
    uint4 u[2];
};

__device__ __forceinline__ unsigned short f32_to_bf16_rne(float f) {
    unsigned u = __float_as_uint(f);
    return (unsigned short)((u + 0x7FFFu + ((u >> 16) & 1u)) >> 16);
}

// ---------------- kernel 0: convert Wb (f32 [o][j]) -> bf16 [o][j] ----------------
__global__ __launch_bounds__(256)
void cvt_wb_bf16(const float* __restrict__ src, unsigned short* __restrict__ dst, int n) {
    int i = blockIdx.x * 256 + threadIdx.x;
    if (i < n) dst[i] = f32_to_bf16_rne(src[i]);
}

// ---------------- kernel 1: a[b][o] = sum_i k[b][i] * Wa[o][i] ----------------
__global__ __launch_bounds__(256)
void proj_k(const float* __restrict__ k, const float* __restrict__ Wa,
            float* __restrict__ aBuf) {
    const int b = blockIdx.y;
    const int o = blockIdx.x * 256 + threadIdx.x;
    const float* kr = k + b * DIM;
    const float* wr = Wa + (size_t)o * DIM;
    float sum = 0.f;
    #pragma unroll 4
    for (int i = 0; i < DIM; ++i) sum += kr[i] * wr[i];
    aBuf[b * DIM + o] = sum;
}

// ---------------- kernel 2: fused scores (WMMA GEMM + tanh + energy dot) ----------
// scores[b][s] = sum_o tanh(a[b][o] + sum_j xs[b][s][j]*Wb[o][j]) * energy[o]
__global__ __launch_bounds__(256)
void fused_scores(const float* __restrict__ xs,
                  const unsigned short* __restrict__ WbBf,   // bf16 bits, [o][j]
                  const float* __restrict__ aBuf,            // [b][o]
                  const float* __restrict__ energy,          // [o]
                  float* __restrict__ scores)                // [b][s]
{
    __shared__ unsigned short Abuf[16 * APAD];   // 16 s-rows x 1024 K (bf16), padded
    __shared__ float sScore[16];

    const int tile = blockIdx.x;                 // 0 .. B*S/16 - 1
    const int b    = tile / (S_LEN / 16);
    const int s0   = (tile % (S_LEN / 16)) * 16;
    const int tid  = threadIdx.x;

    if (tid < 16) sScore[tid] = 0.f;

    // Stage A-tile: 16 rows x 1024 f32 -> bf16 in LDS (read xs exactly once)
    const float* xsTile = xs + ((size_t)b * S_LEN + s0) * DIM;
    for (int idx = tid; idx < 16 * DIM; idx += 256) {
        const int r = idx >> 10, c = idx & (DIM - 1);
        Abuf[r * APAD + c] = f32_to_bf16_rne(xsTile[r * DIM + c]);
    }
    __syncthreads();

    const int  wave = tid >> 5;
    const int  lane = tid & 31;
    const int  m    = lane & 15;      // A: row M; B/C: column N
    const bool hi   = lane >= 16;
    const int  kA   = hi ? 8  : 0;    // A-frag K base per lane half (16-bit A layout)
    const int  kB   = hi ? 8  : 0;    // B-frag K base per lane half (same layout as A)

    float part[8];
    #pragma unroll
    for (int r = 0; r < 8; ++r) part[r] = 0.f;

    // 8 waves split the 64 N-chunks of 16 o-columns each
    for (int ncIdx = 0; ncIdx < 8; ++ncIdx) {
        const int o0 = (wave + ncIdx * 8) * 16;
        const int o  = o0 + m;
        v8f acc = {};
        for (int kc = 0; kc < DIM; kc += 32) {
            FragB16 aF, bF;
            // A: lane holds row m, K = kc+kA+0..7 and kc+kA+16..23 (2x ds_load_b128)
            const uint4* ap = reinterpret_cast<const uint4*>(&Abuf[m * APAD + kc + kA]);
            aF.u[0] = ap[0];
            aF.u[1] = ap[2];
            // B: lane holds column o, K = kc+kB+0..7 and kc+kB+16..23  (B[j][o] = Wb[o][j])
            const uint4* bp = reinterpret_cast<const uint4*>(&WbBf[(size_t)o * DIM + kc + kB]);
            bF.u[0] = bp[0];
            bF.u[1] = bp[2];
            acc = __builtin_amdgcn_wmma_f32_16x16x32_bf16(
                false, aF.h, false, bF.h, (short)0, acc, false, false);
        }
        // C layout: N = lane&15, M = vgpr (+8 for upper lane half)
        const float aVal = aBuf[b * DIM + o];
        const float eVal = energy[o];
        #pragma unroll
        for (int r = 0; r < 8; ++r)
            part[r] += tanhf(acc[r] + aVal) * eVal;
    }

    // Reduce the 16 columns within each lane half (xor masks stay inside the half)
    #pragma unroll
    for (int r = 0; r < 8; ++r) {
        float v = part[r];
        v += __shfl_xor(v, 1, 32);
        v += __shfl_xor(v, 2, 32);
        v += __shfl_xor(v, 4, 32);
        v += __shfl_xor(v, 8, 32);
        part[r] = v;
    }
    if (m == 0) {                       // lanes 0 (rows 0..7) and 16 (rows 8..15)
        const int rowBase = hi ? 8 : 0;
        #pragma unroll
        for (int r = 0; r < 8; ++r) atomicAdd(&sScore[rowBase + r], part[r]);
    }
    __syncthreads();
    if (tid < 16) scores[(size_t)b * S_LEN + s0 + tid] = sScore[tid];
}

// ---------------- kernel 3: masked softmax over S per batch row ----------------
__global__ __launch_bounds__(256)
void masked_softmax(const float* __restrict__ scores,
                    const unsigned char* __restrict__ mask,   // jnp.bool_: 1 byte
                    float* __restrict__ attn) {
    const int b   = blockIdx.x;
    const int tid = threadIdx.x;
    __shared__ float red[256];

    float m = -3.4e38f;
    for (int s = tid; s < S_LEN; s += 256)
        if (mask[b * S_LEN + s]) m = fmaxf(m, scores[b * S_LEN + s]);
    red[tid] = m; __syncthreads();
    for (int off = 128; off > 0; off >>= 1) {
        if (tid < off) red[tid] = fmaxf(red[tid], red[tid + off]);
        __syncthreads();
    }
    m = red[0]; __syncthreads();

    float sum = 0.f;
    for (int s = tid; s < S_LEN; s += 256)
        if (mask[b * S_LEN + s]) sum += __expf(scores[b * S_LEN + s] - m);
    red[tid] = sum; __syncthreads();
    for (int off = 128; off > 0; off >>= 1) {
        if (tid < off) red[tid] += red[tid + off];
        __syncthreads();
    }
    sum = red[0];

    const float inv = 1.f / fmaxf(sum, 1e-30f);
    for (int s = tid; s < S_LEN; s += 256) {
        float v = mask[b * S_LEN + s] ? __expf(scores[b * S_LEN + s] - m) * inv : 0.f;
        attn[b * S_LEN + s] = v;
    }
}

// ---------------- kernel 4: ctx[b][d] = sum_s attn[b][s] * xs[b][s][d] ----------------
__global__ __launch_bounds__(256)
void ctx_reduce(const float* __restrict__ attn, const float* __restrict__ xs,
                float* __restrict__ ctx) {
    const int b = blockIdx.y;
    const int d = blockIdx.x * 256 + threadIdx.x;
    const float* xb = xs + (size_t)b * S_LEN * DIM + d;
    const float* ab = attn + b * S_LEN;
    float sum = 0.f;
    #pragma unroll 4
    for (int s = 0; s < S_LEN; ++s) sum += ab[s] * xb[(size_t)s * DIM];
    ctx[b * DIM + d] = sum;
}

extern "C" void kernel_launch(void* const* d_in, const int* in_sizes, int n_in,
                              void* d_out, int out_size, void* d_ws, size_t ws_size,
                              hipStream_t stream) {
    const float*         k_in   = (const float*)d_in[0];          // [B, DIM]
    const float*         xs     = (const float*)d_in[1];          // [B, S, DIM]
    const unsigned char* mask   = (const unsigned char*)d_in[2];  // [B, S] bool
    const float*         Wa     = (const float*)d_in[3];          // [DIM, DIM]
    const float*         Wb     = (const float*)d_in[4];          // [DIM, DIM]
    const float*         energy = (const float*)d_in[5];          // [DIM]

    float* ctx  = (float*)d_out;                                  // [B, DIM]
    float* attn = (float*)d_out + BB * DIM;                       // [B, S]

    // Workspace layout
    unsigned short* WbBf   = (unsigned short*)d_ws;                               // 2 MB
    float*          aBuf   = (float*)((char*)d_ws + (size_t)DIM * DIM * 2);       // 128 KB
    float*          scores = aBuf + BB * DIM;                                     // 256 KB

    // 0) Wb f32 -> bf16 (stays L2-resident for the GEMM)
    cvt_wb_bf16<<<(DIM * DIM + 255) / 256, 256, 0, stream>>>(Wb, WbBf, DIM * DIM);

    // 1) a = k @ Wa^T
    proj_k<<<dim3(DIM / 256, BB), 256, 0, stream>>>(k_in, Wa, aBuf);

    // 2) fused WMMA scores (dominant: 137 GFLOP through v_wmma_f32_16x16x32_bf16)
    fused_scores<<<(BB * S_LEN) / 16, 256, 0, stream>>>(xs, WbBf, aBuf, energy, scores);

    // 3) masked softmax -> attn
    masked_softmax<<<BB, 256, 0, stream>>>(scores, mask, attn);

    // 4) ctx = attn @ xs (single streaming pass over xs)
    ctx_reduce<<<dim3(DIM / 256, BB), 256, 0, stream>>>(attn, xs, ctx);
}
